// SelfAttentionModule_4621384810681
// MI455X (gfx1250) — matrix operand order repaired and checked by reading, not compiled, not run on hardware
//
#include <hip/hip_runtime.h>
#include <hip/hip_bf16.h>

// ---------------------------------------------------------------------------
// Self-attention (transposed-apply variant) fused for gfx1250 / MI455X.
//   B=4, C=64, CQ=32, HW=4096.
//   energy[i,j] = <q[:,i], k[:,j]>  (K-dim = CQ = 32 -> one v_wmma_f32_16x16x32_f16
//                                    per 16x16 energy tile)
//   attn = softmax_j(energy);  out[c,j] = sum_i v[c,i] attn[i,j]
//   final = Wl*out + bl + x
// Never materializes the 256MB attention matrix. Pass-1 (branch-free, two
// sweeps) computes per-row (max, 1/sum); pass-2 recomputes energy tiles in
// register and streams attn straight into the second WMMA GEMM, fusing the
// Wl projection + bias + residual. All WMMA operands are pre-swizzled in a
// small (~4.3MB, L2-resident) workspace so each lane loads its operand as one
// contiguous 32B v16h. All LDS staging uses packed b128 ops.
// ---------------------------------------------------------------------------

#define HW   4096
#define CC   64
#define CQ   32
#define NTJ  256      // 16-wide tiles along HW

typedef __attribute__((ext_vector_type(16))) _Float16 v16h;
typedef __attribute__((ext_vector_type(8)))  _Float16 v8h;
typedef __attribute__((ext_vector_type(8)))  float    v8f;

// A-matrix 16x32 f16 layout (contiguous k order, matches B):
//   lane = (kl>>4)*16 + m    ;  half = kl & 15
// B-matrix 32x16 f16 layout:
//   lane = (kl>>4)*16 + n    ;  half = kl & 15
__device__ __forceinline__ int a_lane(int m, int kl) { return ((kl >> 4) << 4) + m; }
__device__ __forceinline__ int a_half(int kl)        { return kl & 15; }

__device__ __forceinline__ v8f wmma_f16(v16h a, v16h b, v8f c) {
    return __builtin_amdgcn_wmma_f32_16x16x32_f16(false, a, false, b, (short)0, c, false, false);
}

// ---------------------------------------------------------------------------
// Kernel 1: q = Wq x + bq, k = Wk x + bk; pre-swizzle Q (A-layout of Q^T),
// K (B-layout), V = x (A-layout) into f16 workspace. For a fixed pixel i the
// 32 q (k) values form exactly two contiguous 32B lane-runs -> packed v16h
// stores.
// ---------------------------------------------------------------------------
__global__ __launch_bounds__(256) void pack_qkv(
    const float* __restrict__ x,
    const float* __restrict__ Wq, const float* __restrict__ bq,
    const float* __restrict__ Wk, const float* __restrict__ bk,
    _Float16* __restrict__ Qws, _Float16* __restrict__ Kws,
    _Float16* __restrict__ Vws)
{
    __shared__ float lwq[CQ * CC];
    __shared__ float lwk[CQ * CC];
    __shared__ float lbq[CQ], lbk[CQ];
    for (int t = threadIdx.x; t < CQ * CC; t += 256) { lwq[t] = Wq[t]; lwk[t] = Wk[t]; }
    if (threadIdx.x < CQ) { lbq[threadIdx.x] = bq[threadIdx.x]; lbk[threadIdx.x] = bk[threadIdx.x]; }
    __syncthreads();

    const int pix = blockIdx.x * 256 + threadIdx.x;   // b*HW + i
    const int b   = pix >> 12;
    const int i   = pix & (HW - 1);
    const int m16 = i & 15;

    float q[CQ], k[CQ];
#pragma unroll
    for (int o = 0; o < CQ; ++o) { q[o] = lbq[o]; k[o] = lbk[o]; }

    // V pack: element (m = c&15, kl = i&31) of tile (b, cb = c>>4, kt = i>>5)
    const int vlaneBase = ((i & 31) >> 4) << 4;
    const int vhalf     = a_half(i & 31);
    const long vkt      = i >> 5;

    for (int c = 0; c < CC; ++c) {
        float xv = x[((long)b * CC + c) * HW + i];
        long vtile = ((long)(b * 4 + (c >> 4))) * 128 + vkt;
        Vws[(vtile * 32 + (vlaneBase + (c & 15))) * 16 + vhalf] = (_Float16)xv;
#pragma unroll
        for (int o = 0; o < CQ; ++o) {
            q[o] = fmaf(lwq[o * CC + c], xv, q[o]);
            k[o] = fmaf(lwk[o * CC + c], xv, k[o]);
        }
    }

    const long tQ = ((long)b * NTJ + (i >> 4)) * 32;
    v16h qv0, qv1, kv0, kv1;
#pragma unroll
    for (int h = 0; h < 8; ++h) {
        qv0[h]     = (_Float16)q[h];        // lane m16   , halves 0..7  <- o 0..7
        qv0[h + 8] = (_Float16)q[8 + h];    // lane m16   , halves 8..15 <- o 8..15
        qv1[h]     = (_Float16)q[16 + h];   // lane m16+16, halves 0..7  <- o 16..23
        qv1[h + 8] = (_Float16)q[24 + h];   // lane m16+16, halves 8..15 <- o 24..31
    }
#pragma unroll
    for (int h = 0; h < 16; ++h) {
        kv0[h] = (_Float16)k[h];            // lane m16   , halves 0..15 <- o 0..15
        kv1[h] = (_Float16)k[16 + h];       // lane m16+16, halves 0..15 <- o 16..31
    }
    *(v16h*)(Qws + (tQ + m16) * 16)      = qv0;
    *(v16h*)(Qws + (tQ + 16 + m16) * 16) = qv1;
    *(v16h*)(Kws + (tQ + m16) * 16)      = kv0;
    *(v16h*)(Kws + (tQ + 16 + m16) * 16) = kv1;
}

// ---------------------------------------------------------------------------
// Kernel 1b: pre-swizzle Wl (64x64) into A-layout tiles [ob(4)][ks(2)].
// ---------------------------------------------------------------------------
__global__ __launch_bounds__(256) void pack_wl(const float* __restrict__ Wl,
                                               _Float16* __restrict__ Wlws)
{
    for (int idx = threadIdx.x; idx < CC * CC; idx += 256) {
        int o = idx >> 6, c = idx & 63;
        int tile = ((o >> 4) * 2) + (c >> 5);
        Wlws[((long)tile * 32 + a_lane(o & 15, c & 31)) * 16 + a_half(c & 31)]
            = (_Float16)Wl[idx];
    }
}

// ---------------------------------------------------------------------------
// Kernel 2: per-row softmax stats, branch-free. One wave per i-tile PAIR
// (each streamed K tile feeds 2 WMMAs). Sweep 1: row max. Sweep 2: sum of
// exp with known max. Cross-lane merge via shuffles. Stores (max, 1/sum).
// ---------------------------------------------------------------------------
__global__ __launch_bounds__(128) void row_stats(
    const _Float16* __restrict__ Qws, const _Float16* __restrict__ Kws,
    float2* __restrict__ stats)
{
    const int wid  = threadIdx.x >> 5;
    const int lane = threadIdx.x & 31;
    const int pair = blockIdx.x * 4 + wid;   // b*128 + ip
    const int b    = pair >> 7;
    const int ip   = pair & 127;             // i-tiles 2*ip, 2*ip+1

    const _Float16* qb = Qws + (((long)b * NTJ + ip * 2) * 32) * 16;
    const v16h a0 = *(const v16h*)(qb + lane * 16);
    const v16h a1 = *(const v16h*)(qb + 512 + lane * 16);
    const _Float16* kbase = Kws + (long)b * NTJ * 512;

    float m0[8], m1[8];
#pragma unroll
    for (int r = 0; r < 8; ++r) { m0[r] = -3.0e38f; m1[r] = -3.0e38f; }

    // Sweep 1: max
    for (int u = 0; u < NTJ; ++u) {
        const v16h bt = *(const v16h*)(kbase + (long)u * 512 + lane * 16);
        __builtin_prefetch(kbase + (long)u * 512 + 512 + lane * 16, 0, 1);
        v8f cz = {};
        v8f e0 = wmma_f16(a0, bt, cz);
        v8f e1 = wmma_f16(a1, bt, cz);
#pragma unroll
        for (int r = 0; r < 8; ++r) {
            m0[r] = fmaxf(m0[r], e0[r]);
            m1[r] = fmaxf(m1[r], e1[r]);
        }
    }
#pragma unroll
    for (int mask = 1; mask <= 8; mask <<= 1) {
#pragma unroll
        for (int r = 0; r < 8; ++r) {
            m0[r] = fmaxf(m0[r], __shfl_xor(m0[r], mask, 32));
            m1[r] = fmaxf(m1[r], __shfl_xor(m1[r], mask, 32));
        }
    }

    // Sweep 2: sum of exp(e - max)
    float s0[8], s1[8];
#pragma unroll
    for (int r = 0; r < 8; ++r) { s0[r] = 0.0f; s1[r] = 0.0f; }
    for (int u = 0; u < NTJ; ++u) {
        const v16h bt = *(const v16h*)(kbase + (long)u * 512 + lane * 16);
        __builtin_prefetch(kbase + (long)u * 512 + 512 + lane * 16, 0, 1);
        v8f cz = {};
        v8f e0 = wmma_f16(a0, bt, cz);
        v8f e1 = wmma_f16(a1, bt, cz);
#pragma unroll
        for (int r = 0; r < 8; ++r) {
            s0[r] += __expf(e0[r] - m0[r]);
            s1[r] += __expf(e1[r] - m1[r]);
        }
    }
#pragma unroll
    for (int mask = 1; mask <= 8; mask <<= 1) {
#pragma unroll
        for (int r = 0; r < 8; ++r) {
            s0[r] += __shfl_xor(s0[r], mask, 32);
            s1[r] += __shfl_xor(s1[r], mask, 32);
        }
    }

    if ((lane & 15) == 0) {
        const int hi = lane >> 4;
#pragma unroll
        for (int r = 0; r < 8; ++r) {
            int row0 = (ip * 2) * 16 + r + 8 * hi;
            stats[(long)b * HW + row0]      = make_float2(m0[r], 1.0f / s0[r]);
            stats[(long)b * HW + row0 + 16] = make_float2(m1[r], 1.0f / s1[r]);
        }
    }
}

// ---------------------------------------------------------------------------
// Kernel 3: one wave per j-tile PAIR. Loop over 128 i-pairs: recompute energy
// tiles (2 i x 2 j WMMAs per Q-tile load), normalize to attn in-register,
// stage through LDS (packed v8h = ds b128 stores) into B-layout, accumulate
// out[c,j] with 8 WMMAs (4 c-blocks x 2 j-tiles) per 4 shared V loads.
// Then fuse Wl projection (WMMA via LDS B-operand) + bias + residual.
// ---------------------------------------------------------------------------
__global__ __launch_bounds__(128) void attn_out(
    const _Float16* __restrict__ Qws, const _Float16* __restrict__ Kws,
    const _Float16* __restrict__ Vws, const _Float16* __restrict__ Wlws,
    const float2* __restrict__ stats,
    const float* __restrict__ x, const float* __restrict__ bl,
    float* __restrict__ out)
{
    __shared__ _Float16 smem[4 * 2048];
    const int wid  = threadIdx.x >> 5;
    const int lane = threadIdx.x & 31;
    _Float16* att0 = smem + wid * 2048;          // 16 cols x 32 rows, col-major
    _Float16* att1 = smem + wid * 2048 + 512;
    _Float16* osm  = smem + wid * 2048 + 1024;   // 16 cols x 64 rows, col-major

    const int pairT = blockIdx.x * 4 + wid;      // b*128 + jp
    const int b     = pairT >> 7;
    const int jp    = pairT & 127;               // j-tiles 2*jp, 2*jp+1
    const int j0    = jp * 32;
    const int hi    = lane >> 4;
    const int col   = lane & 15;

    const _Float16* kb = Kws + (((long)b * NTJ + jp * 2) * 32) * 16;
    const v16h kb0 = *(const v16h*)(kb + lane * 16);
    const v16h kb1 = *(const v16h*)(kb + 512 + lane * 16);
    const _Float16* qbase = Qws + (long)b * NTJ * 512;
    const _Float16* vbase = Vws + (long)b * 4 * 128 * 512;
    const float2*   sbase = stats + (long)b * HW;

    v8f acc[2][4];
#pragma unroll
    for (int jj = 0; jj < 2; ++jj)
#pragma unroll
        for (int cb = 0; cb < 4; ++cb) acc[jj][cb] = (v8f){};

    for (int it2 = 0; it2 < 128; ++it2) {
#pragma unroll
        for (int t = 0; t < 2; ++t) {
            const int it = it2 * 2 + t;
            const v16h qa = *(const v16h*)(qbase + (long)it * 512 + lane * 16);
            v8f cz = {};
            v8f e0 = wmma_f16(qa, kb0, cz);
            v8f e1 = wmma_f16(qa, kb1, cz);
            // stats for rows it*16 + 8*hi + 0..7: 64B contiguous -> 4x b128
            const float4* sf4 = (const float4*)(sbase + it * 16 + 8 * hi);
            const float4 sA = sf4[0], sB = sf4[1], sC = sf4[2], sD = sf4[3];
            const float mx[8] = { sA.x, sA.z, sB.x, sB.z, sC.x, sC.z, sD.x, sD.z };
            const float iv[8] = { sA.y, sA.w, sB.y, sB.w, sC.y, sC.w, sD.y, sD.w };
            v8h p0, p1;
#pragma unroll
            for (int r = 0; r < 8; ++r) {
                p0[r] = (_Float16)(__expf(e0[r] - mx[r]) * iv[r]);
                p1[r] = (_Float16)(__expf(e1[r] - mx[r]) * iv[r]);
            }
            // lane's 8 values are contiguous halves -> one ds b128 store each
            *(v8h*)(att0 + col * 32 + t * 16 + 8 * hi) = p0;
            *(v8h*)(att1 + col * 32 + t * 16 + 8 * hi) = p1;
        }
        asm volatile("s_wait_dscnt 0" ::: "memory");
        const v16h ab0 = *(const v16h*)(att0 + col * 32 + hi * 16);
        const v16h ab1 = *(const v16h*)(att1 + col * 32 + hi * 16);
        const v16h v0 = *(const v16h*)(vbase + ((long)(0 * 128 + it2) * 32 + lane) * 16);
        const v16h v1 = *(const v16h*)(vbase + ((long)(1 * 128 + it2) * 32 + lane) * 16);
        const v16h v2 = *(const v16h*)(vbase + ((long)(2 * 128 + it2) * 32 + lane) * 16);
        const v16h v3 = *(const v16h*)(vbase + ((long)(3 * 128 + it2) * 32 + lane) * 16);
        acc[0][0] = wmma_f16(v0, ab0, acc[0][0]);
        acc[0][1] = wmma_f16(v1, ab0, acc[0][1]);
        acc[0][2] = wmma_f16(v2, ab0, acc[0][2]);
        acc[0][3] = wmma_f16(v3, ab0, acc[0][3]);
        acc[1][0] = wmma_f16(v0, ab1, acc[1][0]);
        acc[1][1] = wmma_f16(v1, ab1, acc[1][1]);
        acc[1][2] = wmma_f16(v2, ab1, acc[1][2]);
        acc[1][3] = wmma_f16(v3, ab1, acc[1][3]);
    }

    // Fused Wl projection + bias + residual, one j-tile at a time (osm reused;
    // DS ops from the same wave execute in order, so reuse is safe).
#pragma unroll
    for (int jj = 0; jj < 2; ++jj) {
#pragma unroll
        for (int cb = 0; cb < 4; ++cb) {
            v8h po;
#pragma unroll
            for (int r = 0; r < 8; ++r) po[r] = (_Float16)acc[jj][cb][r];
            *(v8h*)(osm + col * 64 + cb * 16 + 8 * hi) = po;
        }
        asm volatile("s_wait_dscnt 0" ::: "memory");
        const v16h ob0 = *(const v16h*)(osm + col * 64 +      hi * 16);  // c 0..31
        const v16h ob1 = *(const v16h*)(osm + col * 64 + 32 + hi * 16);  // c 32..63
#pragma unroll
        for (int obk = 0; obk < 4; ++obk) {
            const v16h w0 = *(const v16h*)(Wlws + ((long)(obk * 2 + 0) * 32 + lane) * 16);
            const v16h w1 = *(const v16h*)(Wlws + ((long)(obk * 2 + 1) * 32 + lane) * 16);
            v8f cz = {};
            v8f d = wmma_f16(w0, ob0, cz);
            d = wmma_f16(w1, ob1, d);
#pragma unroll
            for (int r = 0; r < 8; ++r) {
                const int o = obk * 16 + r + 8 * hi;
                const long idx = ((long)b * CC + o) * HW + (j0 + jj * 16 + col);
                out[idx] = d[r] + bl[o] + x[idx];
            }
        }
    }
}

// ---------------------------------------------------------------------------
// Launcher (graph-capture safe). Workspace (bytes): Qws[0,1MB) Kws[1,2MB)
// Vws[2,4MB) Wlws[4MB,+8KB) stats[4MB+8KB,+128KB). Total ~4.3MB.
// ---------------------------------------------------------------------------
extern "C" void kernel_launch(void* const* d_in, const int* in_sizes, int n_in,
                              void* d_out, int out_size, void* d_ws, size_t ws_size,
                              hipStream_t stream) {
    const float* x  = (const float*)d_in[0];
    const float* Wq = (const float*)d_in[1];
    const float* bq = (const float*)d_in[2];
    const float* Wk = (const float*)d_in[3];
    const float* bk = (const float*)d_in[4];
    const float* Wl = (const float*)d_in[5];
    const float* bl = (const float*)d_in[6];
    float* out = (float*)d_out;

    char* ws = (char*)d_ws;
    _Float16* Qws   = (_Float16*)(ws);
    _Float16* Kws   = (_Float16*)(ws + (1u << 20));
    _Float16* Vws   = (_Float16*)(ws + (2u << 20));
    _Float16* Wlws  = (_Float16*)(ws + (4u << 20));
    float2*   stats = (float2*)(ws + (4u << 20) + 8192);

    pack_qkv<<<(4 * HW) / 256, 256, 0, stream>>>(x, Wq, bq, Wk, bk, Qws, Kws, Vws);
    pack_wl<<<1, 256, 0, stream>>>(Wl, Wlws);
    row_stats<<<128, 128, 0, stream>>>(Qws, Kws, stats);
    attn_out<<<128, 128, 0, stream>>>(Qws, Kws, Vws, Wlws, stats, x, bl, out);
}
